// OnlineKNN_91156385890953
// MI455X (gfx1250) — matrix operand order repaired and checked by reading, not compiled, not run on hardware
//
#include <hip/hip_runtime.h>
#include <hip/hip_bf16.h>
#include <stddef.h>

// ---- problem constants (from reference) ----
#define BATCH       512
#define FEAT        256
#define QK          65536
#define NUM_KNNS    200
#define NUM_CLASSES 1000
#define INV_TEMP    (1.0f / 0.07f)
#define BINS        8192

typedef __attribute__((ext_vector_type(16))) __bf16 v16bf;
typedef __attribute__((ext_vector_type(8)))  __bf16 v8bf;
typedef __attribute__((ext_vector_type(8)))  float  v8f;

// ---- output layout (flat f32, concatenated in return order) ----
// [0]                       accuracy
// [1 .. 1+QK*FEAT)          new_queue_features
// [.. +QK)                  new_queue_labels (as float values)
// [last]                    new_queue_ptr (as float value)
#define FEAT_OFF   ((size_t)1)
#define LAB_OFF    (FEAT_OFF + (size_t)QK * FEAT)
#define PTR_OFF    (LAB_OFF + (size_t)QK)
#define OUT_TOTAL  (PTR_OFF + 1)

// =====================================================================
// Kernel 1: queue update + output init + fused f32->bf16 conversion of
// the OLD queue and the batch features (GEMM operands). The old queue
// is being streamed for the output copy anyway, so the bf16 conversion
// costs no extra HBM reads; the 32 MB bf16 image stays L2-resident.
// =====================================================================
__global__ void knn_update_kernel(const float* __restrict__ feat,
                                  const int*   __restrict__ labels,
                                  const float* __restrict__ qfeat,
                                  const int*   __restrict__ qlab,
                                  const int*   __restrict__ qptr,
                                  float*  __restrict__ out,
                                  __bf16* __restrict__ qbf,
                                  __bf16* __restrict__ fbf) {
    const int ptr = qptr[0];
    size_t idx    = (size_t)blockIdx.x * blockDim.x + threadIdx.x;
    size_t stride = (size_t)gridDim.x * blockDim.x;
    for (size_t i = idx; i < OUT_TOTAL; i += stride) {
        float val;
        if (i == 0) {
            val = 0.0f;                                   // accuracy accumulator
        } else if (i < LAB_OFF) {
            size_t e = i - FEAT_OFF;
            int row = (int)(e >> 8);                      // e / FEAT
            int col = (int)(e & (FEAT - 1));
            const float old = qfeat[e];
            qbf[e] = (__bf16)old;                         // bf16 OLD queue for GEMM
            if (e < (size_t)BATCH * FEAT)
                fbf[e] = (__bf16)feat[e];                 // bf16 features for GEMM
            int delta = row - ptr; if (delta < 0) delta += QK;
            val = (delta < BATCH) ? feat[(size_t)delta * FEAT + col] : old;
        } else if (i < PTR_OFF) {
            int row = (int)(i - LAB_OFF);
            int delta = row - ptr; if (delta < 0) delta += QK;
            val = (delta < BATCH) ? (float)labels[delta] : (float)qlab[row];
        } else {
            val = (float)((ptr + BATCH) % QK);
        }
        out[i] = val;
    }
}

// =====================================================================
// Kernel 2: sim = feat @ queueT via v_wmma_f32_16x16x32_bf16.
// One wave per 16-column queue tile; B (16 rows x K=256 bf16) held in
// 64 VGPRs, loaded once. Loop over all 32 A row-tiles; hot loop is
// pure b128 bf16 loads + WMMA (no converts).
// =====================================================================
__global__ void __launch_bounds__(128)
knn_gemm_kernel(const __bf16* __restrict__ fbf,
                const __bf16* __restrict__ qbf,
                float* __restrict__ sim) {
    const int lane    = threadIdx.x & 31;
    const int wave    = threadIdx.x >> 5;
    const int colTile = blockIdx.x * 4 + wave;     // 0..4095
    const int n       = lane & 15;                 // B/D column within tile
    const int half    = lane >> 4;                 // lane-half selects K range
    const int m       = lane & 15;                 // A row within tile

    // --- B 32x16 per step: lanes 0-15 hold K=0..15 (pairs per VGPR),
    //     lanes 16-31 hold K=16..31 -> contiguous 32B chunk of queue row.
    const __bf16* qp = qbf + ((size_t)colTile * 16 + n) * FEAT + half * 16;
    v16bf B[8];
#pragma unroll
    for (int s = 0; s < 8; ++s)
        B[s] = *(const v16bf*)(qp + s * 32);

    for (int rt = 0; rt < 32; ++rt) {
        // A 16x32 per step: VGPR0-3 -> K = half*16 + 0..7 (16B chunk),
        //                   VGPR4-7 -> K = half*16 + 8..15 (16B chunk)
        const __bf16* ap = fbf + ((size_t)rt * 16 + m) * FEAT + half * 16;
        v8f acc = {};
#pragma unroll
        for (int s = 0; s < 8; ++s) {
            const v8bf alo = *(const v8bf*)(ap + s * 32);
            const v8bf ahi = *(const v8bf*)(ap + s * 32 + 8);
            v16bf A;
#pragma unroll
            for (int i = 0; i < 8; ++i) { A[i] = alo[i]; A[8 + i] = ahi[i]; }
            acc = __builtin_amdgcn_wmma_f32_16x16x32_bf16(
                false, A, false, B[s], (short)0, acc, false, false);
        }
        // D layout: lane -> N = lane%16, VGPR v -> M = v + 8*(lane/16)
#pragma unroll
        for (int v = 0; v < 8; ++v) {
            const int mm = v + half * 8;
            sim[((size_t)rt * 16 + mm) * QK + (size_t)colTile * 16 + n] = acc[v];
        }
    }
}

// =====================================================================
// Kernel 3: per-row histogram select-k -> boundary bin + allowed count
// =====================================================================
__global__ void __launch_bounds__(256)
knn_thresh_kernel(const float* __restrict__ sim,
                  int* __restrict__ selBin,
                  int* __restrict__ selAllowed) {
    __shared__ unsigned int hist[BINS];
    const int r = blockIdx.x;
    for (int i = threadIdx.x; i < BINS; i += blockDim.x) hist[i] = 0u;
    __syncthreads();

    const float* row = sim + (size_t)r * QK;
    const float lo = -1.05f, scale = (float)BINS / 2.1f;
    for (int k = threadIdx.x; k < QK; k += blockDim.x) {
        int b = (int)((row[k] - lo) * scale);
        b = b < 0 ? 0 : (b > BINS - 1 ? BINS - 1 : b);
        atomicAdd(&hist[b], 1u);
    }
    __syncthreads();

    if (threadIdx.x == 0) {
        unsigned int cum = 0;
        int b = BINS - 1;
        for (; b >= 0; --b) {
            const unsigned int c = hist[b];
            if (cum + c >= NUM_KNNS) {
                selBin[r]     = b;
                selAllowed[r] = (int)(NUM_KNNS - cum);
                break;
            }
            cum += c;
        }
        if (b < 0) { selBin[r] = 0; selAllowed[r] = NUM_KNNS; }
    }
}

// =====================================================================
// Kernel 4: exp-weighted class vote, argmax, accuracy accumulate
// =====================================================================
__global__ void __launch_bounds__(256)
knn_vote_kernel(const float* __restrict__ sim,
                const int*   __restrict__ qlab,
                const int*   __restrict__ labels,
                const int*   __restrict__ selBin,
                const int*   __restrict__ selAllowed,
                float* __restrict__ out_acc) {
    __shared__ float        score[NUM_CLASSES];
    __shared__ unsigned int cnt;
    __shared__ float        bestV[256];
    __shared__ int          bestI[256];

    const int r = blockIdx.x;
    for (int i = threadIdx.x; i < NUM_CLASSES; i += blockDim.x) score[i] = 0.0f;
    if (threadIdx.x == 0) cnt = 0u;
    __syncthreads();

    const int tb      = selBin[r];
    const int allowed = selAllowed[r];
    const float lo = -1.05f, scale = (float)BINS / 2.1f;
    const float* row = sim + (size_t)r * QK;

    for (int k = threadIdx.x; k < QK; k += blockDim.x) {
        const float v = row[k];
        int b = (int)((v - lo) * scale);
        b = b < 0 ? 0 : (b > BINS - 1 ? BINS - 1 : b);
        bool take = (b > tb);
        if (!take && b == tb) {
            const unsigned int o = atomicAdd(&cnt, 1u);
            take = (o < (unsigned int)allowed);
        }
        if (take) atomicAdd(&score[qlab[k]], __expf(v * INV_TEMP));
    }
    __syncthreads();

    // argmax over 1000 classes (prefer lowest index on ties)
    float bv = -1.0f; int bi = NUM_CLASSES;
    for (int c = threadIdx.x; c < NUM_CLASSES; c += blockDim.x) {
        const float s = score[c];
        if (s > bv) { bv = s; bi = c; }
    }
    bestV[threadIdx.x] = bv; bestI[threadIdx.x] = bi;
    __syncthreads();
    for (int off = 128; off > 0; off >>= 1) {
        if (threadIdx.x < off) {
            const float ov = bestV[threadIdx.x + off];
            const int   oi = bestI[threadIdx.x + off];
            if (ov > bestV[threadIdx.x] ||
                (ov == bestV[threadIdx.x] && oi < bestI[threadIdx.x])) {
                bestV[threadIdx.x] = ov; bestI[threadIdx.x] = oi;
            }
        }
        __syncthreads();
    }
    if (threadIdx.x == 0) {
        if (bestI[0] == labels[r]) atomicAdd(out_acc, 1.0f / (float)BATCH);
    }
}

// =====================================================================
extern "C" void kernel_launch(void* const* d_in, const int* in_sizes, int n_in,
                              void* d_out, int out_size, void* d_ws, size_t ws_size,
                              hipStream_t stream) {
    const float* feat   = (const float*)d_in[0];
    const int*   labels = (const int*)  d_in[1];
    const float* qfeat  = (const float*)d_in[2];
    const int*   qlab   = (const int*)  d_in[3];
    const int*   qptr   = (const int*)  d_in[4];
    float*       out    = (float*)d_out;

    // workspace: sim[512*65536] f32 (128 MB) + select state + bf16 operands
    float*  sim        = (float*)d_ws;
    int*    selBin     = (int*)(sim + (size_t)BATCH * QK);
    int*    selAllowed = selBin + BATCH;
    __bf16* qbf        = (__bf16*)(selAllowed + BATCH);        // 65536*256 bf16
    __bf16* fbf        = qbf + (size_t)QK * FEAT;              //   512*256 bf16

    // 1) queue update + output init + bf16 operand materialization
    knn_update_kernel<<<4096, 256, 0, stream>>>(feat, labels, qfeat, qlab, qptr,
                                                out, qbf, fbf);

    // 2) bf16 WMMA GEMM: 4096 column tiles, 4 waves/block
    knn_gemm_kernel<<<1024, 128, 0, stream>>>(fbf, qbf, sim);

    // 3) per-row top-200 threshold via LDS histogram
    knn_thresh_kernel<<<BATCH, 256, 0, stream>>>(sim, selBin, selAllowed);

    // 4) weighted vote + argmax + accuracy
    knn_vote_kernel<<<BATCH, 256, 0, stream>>>(sim, qlab, labels, selBin, selAllowed, out);
}
